// GCNModel_33809982554165
// MI455X (gfx1250) — compile-verified, matrix operand order checked by reading, not run on hardware
//
#include <hip/hip_runtime.h>

// ---------------- problem constants ----------------
#define NQ   30000
#define NSUB 400
#define DD   128
#define HH   128
#define SSEQ 200
#define BBAT 256
#define MSx  4
#define MBx  4
#define PP   512   // H + 3D
#define H3   384   // 3*H
#define D5   640   // 5*D

// ---------------- bf16 / WMMA plumbing ----------------
typedef __bf16 bf16_t;
typedef bf16_t v16bf __attribute__((ext_vector_type(16)));
typedef bf16_t v8bf  __attribute__((ext_vector_type(8)));
typedef float  v8f   __attribute__((ext_vector_type(8)));

static __device__ __forceinline__ unsigned short f2bf_u(float f) {
  union { float f; unsigned u; } v; v.f = f;
  unsigned u = v.u;
  if ((u & 0x7fffffffu) > 0x7f800000u) return (unsigned short)((u >> 16) | 0x40u); // NaN
  u += 0x7fffu + ((u >> 16) & 1u); // round-to-nearest-even
  return (unsigned short)(u >> 16);
}
static __device__ __forceinline__ float bf2f(unsigned short h) {
  union { unsigned u; float f; } v; v.u = ((unsigned)h) << 16; return v.f;
}
static __device__ __forceinline__ float sigmoidf_(float x) { return 1.f / (1.f + expf(-x)); }
static __device__ __forceinline__ v8f v8f_zero() {
  v8f z = {0.f,0.f,0.f,0.f,0.f,0.f,0.f,0.f}; return z;
}

// A/B fragment loader for v_wmma_f32_16x16x32_bf16 (ISA 7.12.2 16-bit layout):
// lane L (L<16): row = base+L, elements 0..7 = K[k..k+7],  8..15 = K[k+16..k+23]
// lane L (>=16): row = base+(L-16), elements 0..7 = K[k+8..k+15], 8..15 = K[k+24..k+31]
static __device__ __forceinline__ v16bf load_frag(const unsigned short* __restrict__ base,
                                                  int row, int ld, int kbase, int lane) {
  const int klo = (lane & 16) ? 8 : 0;
  const unsigned short* p = base + (size_t)row * ld + kbase + klo;
  v8bf lo = *reinterpret_cast<const v8bf*>(p);
  v8bf hi = *reinterpret_cast<const v8bf*>(p + 16);
  return __builtin_shufflevector(lo, hi, 0,1,2,3,4,5,6,7,8,9,10,11,12,13,14,15);
}

static __device__ __forceinline__ v8f wmma_bf16(v16bf a, v16bf b, v8f c) {
  return __builtin_amdgcn_wmma_f32_16x16x32_bf16(false, a, false, b, (short)0, c, false, false);
}

// ---------------- utility kernels ----------------
__global__ void cvt_kernel(const float* __restrict__ src, unsigned short* __restrict__ dst, int n) {
  int i = blockIdx.x * 256 + threadIdx.x;
  if (i < n) dst[i] = f2bf_u(src[i]);
}

__global__ void zero_kernel(float* __restrict__ sums, float* __restrict__ cnts,
                            float* __restrict__ lacc) {
  int i = blockIdx.x * 256 + threadIdx.x;
  if (i < NSUB * DD) sums[i] = 0.f;
  if (i < NSUB)      cnts[i] = 0.f;
  if (i < 2)         lacc[i] = 0.f;
}

__global__ void cnt_kernel(const int* __restrict__ qmap, float* __restrict__ cnts, int n) {
  int i = blockIdx.x * 256 + threadIdx.x;
  if (i < n) atomicAdd(&cnts[qmap[i]], 1.f);
}

// ---------------- generic bf16 WMMA GEMM (16M x 128N per wave) --------------------------
// Used for the Nq-sized GCN GEMMs (M % 16 == 0 but not % 32). grid = (ceil(M/64), N/128)
template<int ACT, bool HAS_CIN, bool OUT_F32>
__global__ void __launch_bounds__(128) gemm_bf16_kernel(
    const unsigned short* __restrict__ A, const unsigned short* __restrict__ W,
    const float* __restrict__ bias, const float* __restrict__ Cin,
    float* __restrict__ outF, unsigned short* __restrict__ outB,
    int M, int N, int K)
{
  const int lane = threadIdx.x & 31;
  const int wave = threadIdx.x >> 5;
  const int m0 = (blockIdx.x * 4 + wave) * 16;
  if (m0 >= M) return;                       // wave-uniform guard (no barriers here)
  const int n0 = blockIdx.y * 128;
  v8f acc[8];
#pragma unroll
  for (int t = 0; t < 8; ++t) acc[t] = v8f_zero();
  const int arow = m0 + (lane & 15);
  const int nrow = n0 + (lane & 15);
  for (int kb = 0; kb < K; kb += 32) {
    v16bf a = load_frag(A, arow, K, kb, lane);
#pragma unroll
    for (int t = 0; t < 8; ++t) {
      v16bf b = load_frag(W, nrow + t * 16, K, kb, lane);
      acc[t] = wmma_bf16(a, b, acc[t]);
    }
  }
  const int mbase = m0 + ((lane >> 4) << 3);
#pragma unroll
  for (int t = 0; t < 8; ++t) {
    const int col = n0 + t * 16 + (lane & 15);
    const float bn = bias ? bias[col] : 0.f;
#pragma unroll
    for (int i = 0; i < 8; ++i) {
      const size_t idx = (size_t)(mbase + i) * (size_t)N + col;
      float v = acc[t][i] + bn;
      if (HAS_CIN) v += Cin[idx];
      if (ACT == 1) v = fmaxf(v, 0.f);
      else if (ACT == 2) v = tanhf(v);
      if (OUT_F32) outF[idx] = v; else outB[idx] = f2bf_u(v);
    }
  }
}

// ---------------- big bf16 WMMA GEMM: 32M x 128N per wave, W staged in LDS --------------
// Block = 4 waves = 128M x 128N.  W k-slab (128 rows x 32 k, 8 KB) is double-buffered in
// LDS and shared by all 4 waves: per 32-K step each wave issues 2 global A-fragment loads
// + 8 ds B-fragment loads for 16 WMMAs.  Requires M % 128 == 0, N % 128 == 0, K % 32 == 0.
// grid = (M/128, N/128)
template<int ACT, bool OUT_F32>
__global__ void __launch_bounds__(128) gemm_bf16_lds_kernel(
    const unsigned short* __restrict__ A, const unsigned short* __restrict__ W,
    const float* __restrict__ bias,
    float* __restrict__ outF, unsigned short* __restrict__ outB,
    int M, int N, int K)
{
  __shared__ __align__(16) unsigned short wtile[2][128 * 32];
  const int tid  = threadIdx.x;
  const int lane = tid & 31;
  const int wave = tid >> 5;
  const int m0w  = blockIdx.x * 128 + wave * 32;
  const int n0   = blockIdx.y * 128;

  v8f acc[2][8];
#pragma unroll
  for (int h = 0; h < 2; ++h)
#pragma unroll
    for (int t = 0; t < 8; ++t) acc[h][t] = v8f_zero();

  const int arow0 = m0w + (lane & 15);
  const int arow1 = arow0 + 16;

  // cooperative stage of one 128x32 W slab into LDS buffer `buf`
  auto stage = [&](int buf, int kb) {
#pragma unroll
    for (int c = 0; c < 4; ++c) {
      const int idx = tid + c * 128;            // 512 16-byte chunks
      const int row = idx >> 2, ch = (idx & 3) * 8;
      const v8bf v = *reinterpret_cast<const v8bf*>(W + (size_t)(n0 + row) * K + kb + ch);
      *reinterpret_cast<v8bf*>(&wtile[buf][row * 32 + ch]) = v;
    }
  };

  stage(0, 0);
  __syncthreads();
  for (int kb = 0; kb < K; kb += 32) {
    const int cur = (kb >> 5) & 1;
    if (kb + 32 < K) stage(cur ^ 1, kb + 32);   // overlap next slab with compute
    // keep the A stream ahead of the k-loop (global_prefetch_b8)
    __builtin_prefetch(A + (size_t)arow0 * K + kb + 64, 0, 1);
    __builtin_prefetch(A + (size_t)arow1 * K + kb + 64, 0, 1);
    v16bf a0 = load_frag(A, arow0, K, kb, lane);
    v16bf a1 = load_frag(A, arow1, K, kb, lane);
#pragma unroll
    for (int t = 0; t < 8; ++t) {
      v16bf b = load_frag(&wtile[cur][0], t * 16 + (lane & 15), 32, 0, lane);
      acc[0][t] = wmma_bf16(a0, b, acc[0][t]);
      acc[1][t] = wmma_bf16(a1, b, acc[1][t]);
    }
    __syncthreads();   // all reads of `cur` done before it is restaged; next slab visible
  }

#pragma unroll
  for (int h = 0; h < 2; ++h) {
    const int mbase = m0w + h * 16 + ((lane >> 4) << 3);
#pragma unroll
    for (int t = 0; t < 8; ++t) {
      const int col = n0 + t * 16 + (lane & 15);
      const float bn = bias ? bias[col] : 0.f;
#pragma unroll
      for (int i = 0; i < 8; ++i) {
        const size_t idx = (size_t)(mbase + i) * (size_t)N + col;
        float v = acc[h][t][i] + bn;
        if (ACT == 1) v = fmaxf(v, 0.f);
        if (OUT_F32) outF[idx] = v; else outB[idx] = f2bf_u(v);
      }
    }
  }
}

// ---------------- q2s projection fused with edge segment-sum scatter ----------------
__global__ void __launch_bounds__(128) q2s_scatter_kernel(
    const unsigned short* __restrict__ A, const unsigned short* __restrict__ W,
    const float* __restrict__ bias, const int* __restrict__ qmap,
    float* __restrict__ sums, int M)
{
  const int lane = threadIdx.x & 31;
  const int wave = threadIdx.x >> 5;
  const int m0 = (blockIdx.x * 4 + wave) * 16;
  if (m0 >= M) return;
  v8f acc[8];
#pragma unroll
  for (int t = 0; t < 8; ++t) acc[t] = v8f_zero();
  const int arow = m0 + (lane & 15);
  const int nrow = lane & 15;
  for (int kb = 0; kb < DD; kb += 32) {
    v16bf a = load_frag(A, arow, DD, kb, lane);
#pragma unroll
    for (int t = 0; t < 8; ++t) {
      v16bf b = load_frag(W, nrow + t * 16, DD, kb, lane);
      acc[t] = wmma_bf16(a, b, acc[t]);
    }
  }
  const int mbase = m0 + ((lane >> 4) << 3);
#pragma unroll
  for (int t = 0; t < 8; ++t) {
    const int col = t * 16 + (lane & 15);
    const float bn = bias[col];
#pragma unroll
    for (int i = 0; i < 8; ++i) {
      const int q = mbase + i;
      const float v = acc[t][i] + bn;
#pragma unroll
      for (int mm = 0; mm < MSx; ++mm) {
        const int s = qmap[q * MSx + mm];
        atomicAdd(&sums[s * DD + col], v);
      }
    }
  }
}

// ---------------- subjects_1 = where(cnt>0, tanh(sums/max(cnt,1) + s_emb@s2s^T + b), 0) ----
__global__ void __launch_bounds__(128) subjects1_kernel(
    const unsigned short* __restrict__ A, const unsigned short* __restrict__ W,
    const float* __restrict__ bias, const float* __restrict__ sums,
    const float* __restrict__ cnts, float* __restrict__ out, int M)
{
  const int lane = threadIdx.x & 31;
  const int wave = threadIdx.x >> 5;
  const int m0 = (blockIdx.x * 4 + wave) * 16;
  if (m0 >= M) return;
  v8f acc[8];
#pragma unroll
  for (int t = 0; t < 8; ++t) acc[t] = v8f_zero();
  const int arow = m0 + (lane & 15);
  const int nrow = lane & 15;
  for (int kb = 0; kb < DD; kb += 32) {
    v16bf a = load_frag(A, arow, DD, kb, lane);
#pragma unroll
    for (int t = 0; t < 8; ++t) {
      v16bf b = load_frag(W, nrow + t * 16, DD, kb, lane);
      acc[t] = wmma_bf16(a, b, acc[t]);
    }
  }
  const int mbase = m0 + ((lane >> 4) << 3);
#pragma unroll
  for (int t = 0; t < 8; ++t) {
    const int col = t * 16 + (lane & 15);
    const float bn = bias[col];
#pragma unroll
    for (int i = 0; i < 8; ++i) {
      const int s = mbase + i;
      const float c = cnts[s];
      float v = acc[t][i] + bn + sums[s * DD + col] / fmaxf(c, 1.f);
      out[s * DD + col] = (c > 0.f) ? tanhf(v) : 0.f;
    }
  }
}

// ---------------- nq = masked mean of subjects_1 over q_map neighbors (bf16 out) -------
__global__ void nq_kernel(const float* __restrict__ s1, const int* __restrict__ qmap,
                          const float* __restrict__ qmask, unsigned short* __restrict__ nq,
                          int total) {
  int i = blockIdx.x * 256 + threadIdx.x;
  if (i >= total) return;
  const int q = i >> 7, d = i & 127;
  float num = 0.f, den = 0.f;
#pragma unroll
  for (int mm = 0; mm < MSx; ++mm) {
    const int s = qmap[q * MSx + mm];
    const float w = qmask[q * MSx + mm];
    num += s1[s * DD + d] * w;
    den += w;
  }
  nq[i] = f2bf_u(den > 0.f ? num / den : 0.f);
}

// ---------------- assemble lstm_input (bf16): [ans|correct|label|question|subjects] -----
__global__ void asm_lstm_kernel(
    const float* __restrict__ ansE, const float* __restrict__ labE,
    const float* __restrict__ q1, const float* __restrict__ s1,
    const int* __restrict__ ans, const int* __restrict__ cans,
    const float* __restrict__ labels, const float* __restrict__ tm,
    const float* __restrict__ vm, const float* __restrict__ lm,
    const int* __restrict__ sids, const float* __restrict__ smask,
    const int* __restrict__ qids, unsigned short* __restrict__ out, int total)
{
  int i = blockIdx.x * 256 + threadIdx.x;
  if (i >= total) return;
  const int d = i % D5;
  const int row = i / D5;                          // row = s*B + b
  float v;
  if (d < 128) {
    const float mask = tm[row] * vm[row] * lm[row];
    v = ansE[ans[row] * DD + d] * mask;
  } else if (d < 256) {
    v = ansE[cans[row] * DD + (d - 128)];
  } else if (d < 384) {
    const float mask = tm[row] * vm[row] * lm[row];
    v = labE[((int)labels[row]) * DD + (d - 256)] * mask;
  } else if (d < 512) {
    v = q1[(size_t)qids[row] * DD + (d - 384)];
  } else {
    float acc = 0.f;
#pragma unroll
    for (int mm = 0; mm < MBx; ++mm)
      acc += s1[sids[row * MBx + mm] * DD + (d - 512)] * smask[row * MBx + mm];
    v = acc;
  }
  out[i] = f2bf_u(v);
}

// ---------------- resident GRU: 16 blocks own 16 batch rows each, loop over S -----------
__global__ void __launch_bounds__(256) gru_kernel(
    const unsigned short* __restrict__ Wh, const float* __restrict__ bh,
    const float* __restrict__ xp, float* __restrict__ fht)
{
  __shared__ __align__(16) float          hf[16 * HH];
  __shared__ __align__(16) unsigned short hb[16 * HH];
  const int tid = threadIdx.x;
  const int lane = tid & 31, wave = tid >> 5;
  const int b0 = blockIdx.x * 16;
  for (int i = tid; i < 16 * HH; i += 256) { hf[i] = 0.f; hb[i] = 0; }

  // preload all Wh B-fragments: wave w owns gate columns [16w,16w+16) of r, z, n
  v16bf whf[3][4];
#pragma unroll
  for (int g = 0; g < 3; ++g) {
    const int nrow = (wave + 8 * g) * 16 + (lane & 15);
#pragma unroll
    for (int kb = 0; kb < 4; ++kb) whf[g][kb] = load_frag(Wh, nrow, HH, kb * 32, lane);
  }
  const int jg = wave * 16 + (lane & 15);
  const float bh0 = bh[jg], bh1 = bh[HH + jg], bh2 = bh[2 * HH + jg];
  const int mb = ((lane >> 4) << 3);

  for (int s = 0; s < SSEQ; ++s) {
    __syncthreads();
    // forward_ht[s] = h (state entering step s)
    const size_t ob = ((size_t)s * BBAT + b0) * HH;
    for (int i = tid; i < 16 * HH; i += 256) fht[ob + i] = hf[i];
    // A fragments from LDS h (bf16), plus h_old and xp gate inputs
    v16bf af[4];
#pragma unroll
    for (int kb = 0; kb < 4; ++kb) af[kb] = load_frag(hb, lane & 15, HH, kb * 32, lane);
    float hold[8], xr[8], xz[8], xn[8];
    const float* xps = xp + ((size_t)s * BBAT + b0) * H3;
#pragma unroll
    for (int i = 0; i < 8; ++i) {
      const int m = mb + i;
      hold[i] = hf[m * HH + jg];
      xr[i] = xps[m * H3 + jg];
      xz[i] = xps[m * H3 + HH + jg];
      xn[i] = xps[m * H3 + 2 * HH + jg];
    }
    __syncthreads();
    v8f a0 = v8f_zero(), a1 = v8f_zero(), a2 = v8f_zero();
#pragma unroll
    for (int kb = 0; kb < 4; ++kb) {
      a0 = wmma_bf16(af[kb], whf[0][kb], a0);
      a1 = wmma_bf16(af[kb], whf[1][kb], a1);
      a2 = wmma_bf16(af[kb], whf[2][kb], a2);
    }
#pragma unroll
    for (int i = 0; i < 8; ++i) {
      const int m = mb + i;
      const float r = sigmoidf_(xr[i] + a0[i] + bh0);
      const float z = sigmoidf_(xz[i] + a1[i] + bh1);
      const float n = tanhf(xn[i] + r * (a2[i] + bh2));
      const float hn = (1.f - z) * n + z * hold[i];
      hf[m * HH + jg] = hn;
      hb[m * HH + jg] = f2bf_u(hn);
    }
  }
}

// ---------------- assemble pred_input (bf16): [question|subjects|forward_ht|correct] ----
__global__ void asm_pred_kernel(const unsigned short* __restrict__ lstm,
                                const float* __restrict__ fht,
                                unsigned short* __restrict__ pred, int total)
{
  int i = blockIdx.x * 256 + threadIdx.x;
  if (i >= total) return;
  const int d = i % PP;
  const size_t row = (size_t)(i / PP);
  unsigned short v;
  if (d < 128)      v = lstm[row * D5 + 384 + d];
  else if (d < 256) v = lstm[row * D5 + 512 + (d - 128)];
  else if (d < 384) v = f2bf_u(fht[row * HH + (d - 256)]);
  else              v = lstm[row * D5 + 128 + (d - 384)];
  pred[i] = v;
}

// ---------------- head: residual out-projection, probs, masked BCE reduction -----------
__global__ void __launch_bounds__(256) final_kernel(
    const unsigned short* __restrict__ h2, const unsigned short* __restrict__ pred,
    const float* __restrict__ outW, const float* __restrict__ outb,
    const float* __restrict__ labels, const float* __restrict__ tm,
    const float* __restrict__ vm, const float* __restrict__ lm,
    float* __restrict__ probs, float* __restrict__ lacc)
{
  const int row = blockIdx.x * 256 + threadIdx.x;   // < S*B
  float acc = 0.f;
  const size_t base = (size_t)row * PP;
  for (int j = 0; j < PP; ++j)
    acc += (bf2f(h2[base + j]) + bf2f(pred[base + j])) * outW[j];
  const float logit = acc + outb[0];
  probs[row] = 1.f / (1.f + expf(-logit));
  const float mask = tm[row] * vm[row] * lm[row];
  const float sp = fmaxf(logit, 0.f) + log1pf(expf(-fabsf(logit)));
  const float per = (sp - logit * labels[row]) * mask;
  __shared__ float rn[256], rd[256];
  rn[threadIdx.x] = per; rd[threadIdx.x] = mask;
  __syncthreads();
  for (int off = 128; off > 0; off >>= 1) {
    if (threadIdx.x < off) { rn[threadIdx.x] += rn[threadIdx.x + off];
                             rd[threadIdx.x] += rd[threadIdx.x + off]; }
    __syncthreads();
  }
  if (threadIdx.x == 0) { atomicAdd(&lacc[0], rn[0]); atomicAdd(&lacc[1], rd[0]); }
}

__global__ void loss_write_kernel(const float* __restrict__ lacc, float* __restrict__ out) {
  out[0] = lacc[1] != 0.f ? lacc[0] / lacc[1] : 0.f;
}

// ---------------- host orchestration ----------------
extern "C" void kernel_launch(void* const* d_in, const int* in_sizes, int n_in,
                              void* d_out, int out_size, void* d_ws, size_t ws_size,
                              hipStream_t stream) {
  const float* q_emb   = (const float*)d_in[0];
  const float* s_emb   = (const float*)d_in[1];
  const float* ans_emb = (const float*)d_in[2];
  const float* lab_emb = (const float*)d_in[3];
  const float* gru_Wi  = (const float*)d_in[4];
  const float* gru_Wh  = (const float*)d_in[5];
  const float* gru_bi  = (const float*)d_in[6];
  const float* gru_bh  = (const float*)d_in[7];
  const float* s2s_W = (const float*)d_in[8];  const float* s2s_b = (const float*)d_in[9];
  const float* s2q_W = (const float*)d_in[10]; const float* s2q_b = (const float*)d_in[11];
  const float* q2q_W = (const float*)d_in[12]; const float* q2q_b = (const float*)d_in[13];
  const float* q2s_W = (const float*)d_in[14]; const float* q2s_b = (const float*)d_in[15];
  const float* l1_W = (const float*)d_in[16];  const float* l1_b = (const float*)d_in[17];
  const float* l2_W = (const float*)d_in[18];  const float* l2_b = (const float*)d_in[19];
  const float* out_W = (const float*)d_in[20]; const float* out_b = (const float*)d_in[21];
  const float* test_mask    = (const float*)d_in[22];
  const float* valid_mask   = (const float*)d_in[23];
  const float* local_mask   = (const float*)d_in[24];
  const float* subject_mask = (const float*)d_in[25];
  const float* q_mask       = (const float*)d_in[26];
  const float* labels       = (const float*)d_in[27];
  const int* ans         = (const int*)d_in[28];
  const int* correct_ans = (const int*)d_in[29];
  const int* subject_ids = (const int*)d_in[30];
  const int* q_ids       = (const int*)d_in[31];
  const int* q_map       = (const int*)d_in[32];

  char* ws = (char*)d_ws;
  size_t off = 0;
  auto alloc = [&](size_t bytes) -> char* {
    off = (off + 255) & ~(size_t)255;
    char* p = ws + off; off += bytes; return p;
  };
  unsigned short* q2sW_b = (unsigned short*)alloc((size_t)DD * DD * 2);
  unsigned short* s2sW_b = (unsigned short*)alloc((size_t)DD * DD * 2);
  unsigned short* s2qW_b = (unsigned short*)alloc((size_t)DD * DD * 2);
  unsigned short* q2qW_b = (unsigned short*)alloc((size_t)DD * DD * 2);
  unsigned short* gWi_b  = (unsigned short*)alloc((size_t)H3 * D5 * 2);
  unsigned short* gWh_b  = (unsigned short*)alloc((size_t)H3 * HH * 2);
  unsigned short* l1W_b  = (unsigned short*)alloc((size_t)PP * PP * 2);
  unsigned short* l2W_b  = (unsigned short*)alloc((size_t)PP * PP * 2);
  unsigned short* qemb_b = (unsigned short*)alloc((size_t)NQ * DD * 2);
  unsigned short* semb_b = (unsigned short*)alloc((size_t)NSUB * DD * 2);
  float* sums  = (float*)alloc((size_t)NSUB * DD * 4);
  float* cnts  = (float*)alloc((size_t)NSUB * 4);
  float* subj1 = (float*)alloc((size_t)NSUB * DD * 4);
  unsigned short* nq_b = (unsigned short*)alloc((size_t)NQ * DD * 2);
  float* tmpq  = (float*)alloc((size_t)NQ * DD * 4);
  float* ques1 = (float*)alloc((size_t)NQ * DD * 4);
  unsigned short* lstm_b = (unsigned short*)alloc((size_t)SSEQ * BBAT * D5 * 2);
  float* xp    = (float*)alloc((size_t)SSEQ * BBAT * H3 * 4);
  float* fht   = (float*)alloc((size_t)SSEQ * BBAT * HH * 4);
  unsigned short* pred_b = (unsigned short*)alloc((size_t)SSEQ * BBAT * PP * 2);
  unsigned short* h1_b   = (unsigned short*)alloc((size_t)SSEQ * BBAT * PP * 2);
  unsigned short* h2_b   = (unsigned short*)alloc((size_t)SSEQ * BBAT * PP * 2);
  float* lacc  = (float*)alloc(2 * 4);
  (void)ws_size; (void)in_sizes; (void)n_in; (void)out_size;

  float* out_f = (float*)d_out;           // [0] = loss, [1..] = probs (S*B)

  auto cvt = [&](const float* s, unsigned short* d, int n) {
    cvt_kernel<<<(n + 255) / 256, 256, 0, stream>>>(s, d, n);
  };

  zero_kernel<<<(NSUB * DD + 255) / 256, 256, 0, stream>>>(sums, cnts, lacc);

  cvt(q2s_W, q2sW_b, DD * DD);   cvt(s2s_W, s2sW_b, DD * DD);
  cvt(s2q_W, s2qW_b, DD * DD);   cvt(q2q_W, q2qW_b, DD * DD);
  cvt(gru_Wi, gWi_b, H3 * D5);   cvt(gru_Wh, gWh_b, H3 * HH);
  cvt(l1_W, l1W_b, PP * PP);     cvt(l2_W, l2W_b, PP * PP);
  cvt(q_emb, qemb_b, NQ * DD);   cvt(s_emb, semb_b, NSUB * DD);

  // edge counts (segment_sum of ones over q_map)
  cnt_kernel<<<(NQ * MSx + 255) / 256, 256, 0, stream>>>(q_map, cnts, NQ * MSx);

  auto gblocks = [](int M) { return (M / 16 + 3) / 4; };

  // GCN layer 1: q2s projection + scatter into sums
  q2s_scatter_kernel<<<dim3(gblocks(NQ), 1), 128, 0, stream>>>(
      qemb_b, q2sW_b, q2s_b, q_map, sums, NQ);
  subjects1_kernel<<<dim3(gblocks(NSUB), 1), 128, 0, stream>>>(
      semb_b, s2sW_b, s2s_b, sums, cnts, subj1, NSUB);

  // GCN layer 2
  nq_kernel<<<(NQ * DD + 255) / 256, 256, 0, stream>>>(subj1, q_map, q_mask, nq_b, NQ * DD);
  gemm_bf16_kernel<0, false, true><<<dim3(gblocks(NQ), 1), 128, 0, stream>>>(
      qemb_b, q2qW_b, q2q_b, nullptr, tmpq, nullptr, NQ, DD, DD);
  gemm_bf16_kernel<2, true, true><<<dim3(gblocks(NQ), 1), 128, 0, stream>>>(
      nq_b, s2qW_b, s2q_b, tmpq, ques1, nullptr, NQ, DD, DD);

  // sequence feature assembly + input GEMM (LDS-staged, 32M/wave)
  {
    const int total = SSEQ * BBAT * D5;
    asm_lstm_kernel<<<(total + 255) / 256, 256, 0, stream>>>(
        ans_emb, lab_emb, ques1, subj1, ans, correct_ans, labels,
        test_mask, valid_mask, local_mask, subject_ids, subject_mask, q_ids,
        lstm_b, total);
  }
  gemm_bf16_lds_kernel<0, true><<<dim3(SSEQ * BBAT / 128, H3 / 128), 128, 0, stream>>>(
      lstm_b, gWi_b, gru_bi, xp, nullptr, SSEQ * BBAT, H3, D5);

  // resident GRU over S=200 steps
  gru_kernel<<<BBAT / 16, 256, 0, stream>>>(gWh_b, gru_bh, xp, fht);

  // prediction head (LDS-staged GEMMs)
  {
    const int total = SSEQ * BBAT * PP;
    asm_pred_kernel<<<(total + 255) / 256, 256, 0, stream>>>(lstm_b, fht, pred_b, total);
  }
  gemm_bf16_lds_kernel<1, false><<<dim3(SSEQ * BBAT / 128, PP / 128), 128, 0, stream>>>(
      pred_b, l1W_b, l1_b, nullptr, h1_b, SSEQ * BBAT, PP, PP);
  gemm_bf16_lds_kernel<1, false><<<dim3(SSEQ * BBAT / 128, PP / 128), 128, 0, stream>>>(
      h1_b, l2W_b, l2_b, nullptr, h2_b, SSEQ * BBAT, PP, PP);

  final_kernel<<<SSEQ * BBAT / 256, 256, 0, stream>>>(
      h2_b, pred_b, out_W, out_b, labels, test_mask, valid_mask, local_mask,
      out_f + 1, lacc);
  loss_write_kernel<<<1, 1, 0, stream>>>(lacc, out_f);
}
